// FeatureNet_4535485464644
// MI455X (gfx1250) — matrix operand order repaired and checked by reading, not compiled, not run on hardware
//
#include <hip/hip_runtime.h>

#define BB 8
#define NN 4096
#define KK 8
#define DD 128

typedef __attribute__((ext_vector_type(16))) _Float16 v16h;
typedef __attribute__((ext_vector_type(8)))  float    v8f;

__device__ __forceinline__ v8f zero8() {
  v8f z;
  #pragma unroll
  for (int i = 0; i < 8; ++i) z[i] = 0.0f;
  return z;
}
__device__ __forceinline__ v16h zero16h() {
  v16h z;
  #pragma unroll
  for (int i = 0; i < 16; ++i) z[i] = (_Float16)0.0f;
  return z;
}

// ---------------------------------------------------------------------------
// Kernel 0: pre-swizzle weights into WMMA B-operand layout (f16).
// B layout for v_wmma_f32_16x16x32_f16 (per CDNA5 ISA 7.12): N = lane&15,
// packed half e (0..15): K = (e<8? e : e+8) + 8*(L>=16).  Stored so lane L reads 16 contiguous halfs:
//   wXh[((t*4 + kc)*32 + L)*16 + e] = W[(t*16 + (L&15))*128 + kc*32 + K(L,e)]
// W0 (128x3) gets one K-chunk, K >= 3 zero-padded.
// ---------------------------------------------------------------------------
__global__ void prep_weights(const float* __restrict__ W0,
                             const float* __restrict__ W1,
                             const float* __restrict__ W2,
                             _Float16* __restrict__ w0h,
                             _Float16* __restrict__ w1h,
                             _Float16* __restrict__ w2h) {
  int i  = blockIdx.x * blockDim.x + threadIdx.x;   // 0..16383
  int e  = i & 15;
  int L  = (i >> 4) & 31;
  int kc = (i >> 9) & 3;
  int t  = i >> 11;
  int kk = (e < 8 ? e : e + 8) + ((L >> 4) << 3);
  w1h[i] = (_Float16)W1[(t * 16 + (L & 15)) * DD + kc * 32 + kk];
  w2h[i] = (_Float16)W2[(t * 16 + (L & 15)) * DD + kc * 32 + kk];
  if (i < 4096) {
    int t0 = i >> 9;                                // 0..7
    w0h[i] = (kk < 3) ? (_Float16)W0[(t0 * 16 + (L & 15)) * 3 + kk] : (_Float16)0.0f;
  }
}

// ---------------------------------------------------------------------------
// Kernel 1: brute-force KNN (top-8, self excluded).  One thread per query
// point; candidate points tiled through LDS; sorted top-8 in registers with a
// strict-< bubble insert (matches top_k tie-break by lower index).
// ---------------------------------------------------------------------------
__global__ void knn_kernel(const float* __restrict__ x, int* __restrict__ idxb) {
  __shared__ float cxs[128], cys[128], czs[128], cqs[128];
  int b = blockIdx.y;
  int n = blockIdx.x * 128 + threadIdx.x;
  const float* xb = x + b * 3 * NN;
  float qx = xb[n], qy = xb[NN + n], qz = xb[2 * NN + n];
  float qsq = qx * qx + qy * qy + qz * qz;

  float bd[KK];
  int   bi[KK];
  #pragma unroll
  for (int k = 0; k < KK; ++k) { bd[k] = 3.4e38f; bi[k] = 0; }

  for (int j0 = 0; j0 < NN; j0 += 128) {
    __syncthreads();
    int jl = j0 + threadIdx.x;
    float tx = xb[jl], ty = xb[NN + jl], tz = xb[2 * NN + jl];
    cxs[threadIdx.x] = tx;
    cys[threadIdx.x] = ty;
    czs[threadIdx.x] = tz;
    cqs[threadIdx.x] = tx * tx + ty * ty + tz * tz;
    __syncthreads();
    for (int jj = 0; jj < 128; ++jj) {
      int j = j0 + jj;
      float d2 = qsq + cqs[jj] - 2.0f * (qx * cxs[jj] + qy * cys[jj] + qz * czs[jj]);
      if (j != n && d2 < bd[KK - 1]) {
        bd[KK - 1] = d2;
        bi[KK - 1] = j;
        #pragma unroll
        for (int s = KK - 1; s > 0; --s) {
          if (bd[s] < bd[s - 1]) {
            float td = bd[s]; bd[s] = bd[s - 1]; bd[s - 1] = td;
            int   ti = bi[s]; bi[s] = bi[s - 1]; bi[s - 1] = ti;
          }
        }
      }
    }
  }
  #pragma unroll
  for (int k = 0; k < KK; ++k) idxb[(b * NN + n) * KK + k] = bi[k];
}

// ---------------------------------------------------------------------------
// Kernel 2: fused gather + 3-layer ReLU MLP (WMMA f16->f32) + max over K.
// One wave (32 threads) handles 16 rows = 2 points x 8 neighbors.
// A layout (f16 16x32): lane = M (&15), halfs e: K = (e<8? e : 16+e-8) + 8*(lane>=16).
// C/D layout (f32 16x16): VGPR r, lane L -> row r + 8*(L>=16), col L&15.
// Layer transitions transpose C-layout -> A-layout through LDS (ReLU fused).
// ---------------------------------------------------------------------------
__global__ void __launch_bounds__(32) mlp_kernel(const float* __restrict__ x,
                                                 const int* __restrict__ idxb,
                                                 const _Float16* __restrict__ w0h,
                                                 const _Float16* __restrict__ w1h,
                                                 const _Float16* __restrict__ w2h,
                                                 float* __restrict__ out) {
  __shared__ float hbuf[16 * DD];
  int lane = threadIdx.x;
  int b    = blockIdx.y;
  int n0   = blockIdx.x * 2;
  int row  = lane & 15;
  int g    = lane >> 4;
  int g8   = g * 8;
  const float* xb = x + b * 3 * NN;

  // ---- layer 0 A operand: rel (16x3), zero padded to K=32 ----
  int p = row >> 3, k = row & 7;
  int n = n0 + p;
  int j = idxb[(b * NN + n) * KK + k];
  float rx = xb[j]          - xb[n];
  float ry = xb[NN + j]     - xb[NN + n];
  float rz = xb[2 * NN + j] - xb[2 * NN + n];
  v16h a0 = zero16h();
  if (lane < 16) {             // lanes 16..31 hold K>=8 halves -> all zero
    a0[0] = (_Float16)rx;
    a0[1] = (_Float16)ry;
    a0[2] = (_Float16)rz;
  }

  v8f acc[8];
  #pragma unroll
  for (int t = 0; t < 8; ++t) {
    v16h bmat = *(const v16h*)(w0h + (t * 32 + lane) * 16);
    acc[t] = __builtin_amdgcn_wmma_f32_16x16x32_f16(false, a0, false, bmat,
                                                    (short)0, zero8(), false, false);
  }

  const _Float16* wl[2] = { w1h, w2h };
  #pragma unroll
  for (int layer = 0; layer < 2; ++layer) {
    __syncthreads();
    #pragma unroll
    for (int t = 0; t < 8; ++t)
      #pragma unroll
      for (int r = 0; r < 8; ++r)
        hbuf[(r + g8) * DD + t * 16 + row] = fmaxf(acc[t][r], 0.0f);   // ReLU fused
    __syncthreads();

    const float* hrow = hbuf + row * DD;
    v16h A[4];
    #pragma unroll
    for (int kc = 0; kc < 4; ++kc) {
      #pragma unroll
      for (int e = 0; e < 8; ++e) A[kc][e]     = (_Float16)hrow[kc * 32 + g8 + e];
      #pragma unroll
      for (int e = 0; e < 8; ++e) A[kc][8 + e] = (_Float16)hrow[kc * 32 + 16 + g8 + e];
    }

    const _Float16* w = wl[layer];
    #pragma unroll
    for (int t = 0; t < 8; ++t) {
      v8f c = zero8();
      #pragma unroll
      for (int kc = 0; kc < 4; ++kc) {
        v16h bmat = *(const v16h*)(w + ((t * 4 + kc) * 32 + lane) * 16);
        c = __builtin_amdgcn_wmma_f32_16x16x32_f16(false, A[kc], false, bmat,
                                                   (short)0, c, false, false);
      }
      acc[t] = c;
    }
  }

  // ---- ReLU + max over the 8 neighbors (per-lane over acc components) ----
  int nout = n0 + g;
  #pragma unroll
  for (int t = 0; t < 8; ++t) {
    float m = acc[t][0];
    #pragma unroll
    for (int r = 1; r < 8; ++r) m = fmaxf(m, acc[t][r]);
    m = fmaxf(m, 0.0f);
    out[(b * DD + (t * 16 + row)) * NN + nout] = m;   // out shape (B,128,N)
  }
}

// ---------------------------------------------------------------------------
extern "C" void kernel_launch(void* const* d_in, const int* in_sizes, int n_in,
                              void* d_out, int out_size, void* d_ws, size_t ws_size,
                              hipStream_t stream) {
  (void)in_sizes; (void)n_in; (void)out_size; (void)ws_size;
  const float* x  = (const float*)d_in[0];
  const float* W0 = (const float*)d_in[1];
  const float* W1 = (const float*)d_in[2];
  const float* W2 = (const float*)d_in[3];
  float* out = (float*)d_out;

  char* ws = (char*)d_ws;
  size_t off = 0;
  int* idxb = (int*)(ws + off);            off += (size_t)BB * NN * KK * sizeof(int); // 1 MiB
  _Float16* w0h = (_Float16*)(ws + off);   off += 4096 * sizeof(_Float16);            // 8 KiB
  _Float16* w1h = (_Float16*)(ws + off);   off += 16384 * sizeof(_Float16);           // 32 KiB
  _Float16* w2h = (_Float16*)(ws + off);

  prep_weights<<<dim3(64), dim3(256), 0, stream>>>(W0, W1, W2, w0h, w1h, w2h);
  knn_kernel<<<dim3(NN / 128, BB), dim3(128), 0, stream>>>(x, idxb);
  mlp_kernel<<<dim3(NN / 2, BB), dim3(32), 0, stream>>>(x, idxb, w0h, w1h, w2h, out);
}
